// Attention_79620103733970
// MI455X (gfx1250) — matrix operand order repaired and checked by reading, not compiled, not run on hardware
//
#include <hip/hip_runtime.h>
#include <hip/hip_bf16.h>
#include <math.h>

// Problem constants (match the reference).
#define Bb   2
#define Tdim 2048
#define Cdim 1024
#define Hh   16
#define HSd  64

typedef __attribute__((ext_vector_type(16))) __bf16 bf16x16;
typedef __attribute__((ext_vector_type(8)))  float  f32x8;

// ---------------------------------------------------------------------------
// CDNA5 async-to-LDS path (ASYNCcnt-tracked, no VGPR staging).
// ---------------------------------------------------------------------------
#if defined(__has_builtin)
#if __has_builtin(__builtin_amdgcn_global_load_async_to_lds_b128) && \
    __has_builtin(__builtin_amdgcn_s_wait_asynccnt)
#define HAS_ASYNC 1
#endif
#if __has_builtin(__builtin_amdgcn_sched_barrier)
#define SCHED_BARRIER() __builtin_amdgcn_sched_barrier(0)
#endif
#endif
#ifndef HAS_ASYNC
#define HAS_ASYNC 0
#endif
#ifndef SCHED_BARRIER
#define SCHED_BARRIER()
#endif

#if HAS_ASYNC
typedef __attribute__((__vector_size__(4 * sizeof(int)))) int v4i_;
typedef __attribute__((address_space(1))) v4i_* gv4i_p;   // global int4*
typedef __attribute__((address_space(3))) v4i_* lv4i_p;   // LDS int4*

__device__ __forceinline__ void async_copy_b128(void* lds, const void* g) {
  __builtin_amdgcn_global_load_async_to_lds_b128(
      (gv4i_p)g, (lv4i_p)lds, 0, 0);
}
#endif

// ---------------------------------------------------------------------------
// helpers
// ---------------------------------------------------------------------------
__device__ __forceinline__ f32x8 wmma_bf16(bf16x16 a, bf16x16 b, f32x8 c) {
  return __builtin_amdgcn_wmma_f32_16x16x32_bf16(
      false, a, false, b, (short)0, c, false, false);
}

// A-fragment (16-bit A layout): per lane, two disjoint 16B chunks.
__device__ __forceinline__ bf16x16 load_frag_split(const __bf16* lo,
                                                   const __bf16* hi) {
  bf16x16 f;
  __builtin_memcpy(&f, lo, 16);
  __builtin_memcpy(reinterpret_cast<char*>(&f) + 16, hi, 16);
  return f;
}

// B-fragment: per lane, 16 contiguous K values for its column (32 bytes).
__device__ __forceinline__ bf16x16 load_frag32(const __bf16* p) {
  bf16x16 f;
  __builtin_memcpy(&f, p, 32);
  return f;
}

__device__ __forceinline__ unsigned pack_bf2(float x, float y) {
  union { __bf16 h[2]; unsigned u; } t;
  t.h[0] = (__bf16)x; t.h[1] = (__bf16)y;
  return t.u;
}

// ---------------------------------------------------------------------------
// Shared 128x64 GEMM mainloop (bf16 WMMA, f32 accum, double-buffered LDS).
//  4 waves; each wave owns a 32x64 tile (2 A-frags x 4 B-frags -> 8 WMMAs).
//  A: [*, Cdim] f32 row-major.  W: [Cdim, Cdim] f32 row-major.
// ---------------------------------------------------------------------------
#define LDA 48   // As row stride (bf16): 128 rows x 32 k (96B, 16B aligned)
#define LDB 48   // Bs row stride (bf16): 64 n     x 32 k (column-major B)

__device__ __forceinline__ void gemm128x64_bf16(
    const float* __restrict__ A, const float* __restrict__ W,
    int rowBase, int colBase,
    __bf16 (*As)[128 * LDA], __bf16 (*Bs)[64 * LDB], f32x8 acc[2][4]) {
  const int tid  = threadIdx.x;
  const int lane = tid & 31;
  const int wave = tid >> 5;
  const int hf = lane >> 4, ln = lane & 15;
  const int arow = tid;                              // A stage: 128r x 32k
  const int bk_  = tid >> 2, bn0 = (tid & 3) * 16;   // B stage: 32k x 64n

  float4 ar4[8], br4[4];

  // Prologue: stage k0 = 0 into buffer 0.
  {
    const float4* s4 = reinterpret_cast<const float4*>(
        A + (size_t)(rowBase + arow) * Cdim);
    #pragma unroll
    for (int j = 0; j < 8; ++j) ar4[j] = s4[j];
    const float4* t4 = reinterpret_cast<const float4*>(
        W + (size_t)bk_ * Cdim + colBase + bn0);
    #pragma unroll
    for (int j = 0; j < 4; ++j) br4[j] = t4[j];
    #pragma unroll
    for (int j = 0; j < 8; ++j) {
      *(unsigned*)&As[0][arow * LDA + 4 * j]     = pack_bf2(ar4[j].x, ar4[j].y);
      *(unsigned*)&As[0][arow * LDA + 4 * j + 2] = pack_bf2(ar4[j].z, ar4[j].w);
    }
    #pragma unroll
    for (int j = 0; j < 4; ++j) {
      Bs[0][(bn0 + 4 * j + 0) * LDB + bk_] = (__bf16)br4[j].x;
      Bs[0][(bn0 + 4 * j + 1) * LDB + bk_] = (__bf16)br4[j].y;
      Bs[0][(bn0 + 4 * j + 2) * LDB + bk_] = (__bf16)br4[j].z;
      Bs[0][(bn0 + 4 * j + 3) * LDB + bk_] = (__bf16)br4[j].w;
    }
  }

  int cur = 0;
  for (int k0 = 0; k0 < Cdim; k0 += 32) {
    __syncthreads();                       // buf[cur] ready, prior reads done
    const bool has_next = (k0 + 32) < Cdim;
    if (has_next) {                        // issue next tile's global loads now
      const float4* s4 = reinterpret_cast<const float4*>(
          A + (size_t)(rowBase + arow) * Cdim + k0 + 32);
      #pragma unroll
      for (int j = 0; j < 8; ++j) ar4[j] = s4[j];
      const float4* t4 = reinterpret_cast<const float4*>(
          W + (size_t)(k0 + 32 + bk_) * Cdim + colBase + bn0);
      #pragma unroll
      for (int j = 0; j < 4; ++j) br4[j] = t4[j];
    }
    // Compute on buf[cur]: all 6 frag loads first, then 8 WMMAs.
    const __bf16* ar0 = &As[cur][(32 * wave + ln) * LDA];
    const __bf16* ar1 = &As[cur][(32 * wave + 16 + ln) * LDA];
    bf16x16 af[2];
    af[0] = load_frag_split(ar0 + 16 * hf, ar0 + 16 * hf + 8);
    af[1] = load_frag_split(ar1 + 16 * hf, ar1 + 16 * hf + 8);
    bf16x16 bfr[4];
    #pragma unroll
    for (int nt = 0; nt < 4; ++nt)
      bfr[nt] = load_frag32(&Bs[cur][(16 * nt + ln) * LDB + 16 * hf]);
    SCHED_BARRIER();                       // keep DS loads ahead of WMMAs
    #pragma unroll
    for (int nt = 0; nt < 4; ++nt)
      acc[0][nt] = wmma_bf16(af[0], bfr[nt], acc[0][nt]);
    #pragma unroll
    for (int nt = 0; nt < 4; ++nt)
      acc[1][nt] = wmma_bf16(af[1], bfr[nt], acc[1][nt]);
    SCHED_BARRIER();
    // Convert + store next tile behind the WMMAs.
    if (has_next) {
      const int nb = cur ^ 1;
      #pragma unroll
      for (int j = 0; j < 8; ++j) {
        *(unsigned*)&As[nb][arow * LDA + 4 * j]     = pack_bf2(ar4[j].x, ar4[j].y);
        *(unsigned*)&As[nb][arow * LDA + 4 * j + 2] = pack_bf2(ar4[j].z, ar4[j].w);
      }
      #pragma unroll
      for (int j = 0; j < 4; ++j) {
        Bs[nb][(bn0 + 4 * j + 0) * LDB + bk_] = (__bf16)br4[j].x;
        Bs[nb][(bn0 + 4 * j + 1) * LDB + bk_] = (__bf16)br4[j].y;
        Bs[nb][(bn0 + 4 * j + 2) * LDB + bk_] = (__bf16)br4[j].z;
        Bs[nb][(bn0 + 4 * j + 3) * LDB + bk_] = (__bf16)br4[j].w;
      }
    }
    cur ^= 1;
  }
}

// ---------------------------------------------------------------------------
// Kernel 1: fused QKV projection + bias + RoPE.
//  grid = (M/128, 48); blockIdx.y selects {q,k,v} x 16 column tiles.
//  Output [B, H, T, HS] bf16.
// ---------------------------------------------------------------------------
__global__ __launch_bounds__(128) void qkv_rope_kernel(
    const float* __restrict__ x,
    const float* __restrict__ wq, const float* __restrict__ wk,
    const float* __restrict__ wv,
    const float* __restrict__ bq, const float* __restrict__ bk,
    const float* __restrict__ bv,
    __bf16* __restrict__ qo, __bf16* __restrict__ ko, __bf16* __restrict__ vo) {
  __shared__ __bf16 As[2][128 * LDA];
  __shared__ __bf16 Bs[2][64 * LDB];

  const int tid  = threadIdx.x;
  const int lane = tid & 31;
  const int wave = tid >> 5;
  const int hf = lane >> 4, ln = lane & 15;

  const int rowBase = blockIdx.x * 128;
  const int nTile   = blockIdx.y;
  const int which   = nTile >> 4;             // 0=q 1=k 2=v
  const int colBase = (nTile & 15) * 64;

  const float* w    = (which == 0) ? wq : (which == 1) ? wk : wv;
  const float* bias = (which == 0) ? bq : (which == 1) ? bk : bv;
  __bf16*      outp = (which == 0) ? qo : (which == 1) ? ko : vo;

  f32x8 acc[2][4] = {};
  gemm128x64_bf16(x, w, rowBase, colBase, As, Bs, acc);

  // bias
  #pragma unroll
  for (int rt = 0; rt < 2; ++rt)
    #pragma unroll
    for (int nt = 0; nt < 4; ++nt)
      #pragma unroll
      for (int r = 0; r < 8; ++r)
        acc[rt][nt][r] += bias[colBase + 16 * nt + ln];

  // RoPE: d <-> d+32 pairs live in accumulators nt <-> nt+2 (same lane/VGPR).
  const int h = colBase >> 6;
  if (which < 2) {
    #pragma unroll
    for (int nt = 0; nt < 2; ++nt) {
      const int d = 16 * nt + ln;                                // 0..31
      const float inv_freq = __expf(-(float)d * 0.2878231366f); // 1e4^(-d/32)
      #pragma unroll
      for (int rt = 0; rt < 2; ++rt)
        #pragma unroll
        for (int r = 0; r < 8; ++r) {
          const int row = rowBase + 32 * wave + 16 * rt + r + 8 * hf;
          const float t = (float)(row & (Tdim - 1));
          float s, c;
          __sincosf(t * inv_freq, &s, &c);
          const float x1 = acc[rt][nt][r], x2 = acc[rt][nt + 2][r];
          acc[rt][nt][r]     = x1 * c - x2 * s;
          acc[rt][nt + 2][r] = x2 * c + x1 * s;
        }
    }
  }

  #pragma unroll
  for (int rt = 0; rt < 2; ++rt)
    #pragma unroll
    for (int nt = 0; nt < 4; ++nt)
      #pragma unroll
      for (int r = 0; r < 8; ++r) {
        const int row = rowBase + 32 * wave + 16 * rt + r + 8 * hf;
        const int t = row & (Tdim - 1), b = row >> 11;
        const int d = 16 * nt + ln;
        outp[(((size_t)b * Hh + h) * Tdim + t) * HSd + d] = (__bf16)acc[rt][nt][r];
      }
}

// ---------------------------------------------------------------------------
// Kernel 2: causal flash attention over bf16 Q/K/V in [B,H,T,HS].
//  grid = (T/64, B*H); 4 waves; double-buffered K/V tiles (K via async LDS
//  DMA); one barrier per 32-key tile.
// ---------------------------------------------------------------------------
#define LKT 72   // K  tile row stride: 32 keys x 64 d  (144B, conflict-free)
#define LVT 40   // V^T tile row stride: 64 d  x 32 keys (80B)
#define LPT 40   // P  patch row stride: 16 q  x 32 keys

__global__ __launch_bounds__(128) void attn_kernel(
    const __bf16* __restrict__ Q, const __bf16* __restrict__ K,
    const __bf16* __restrict__ V, float* __restrict__ Y) {
  __shared__ __bf16 Kt[2][32 * LKT];
  __shared__ __bf16 Vt[2][64 * LVT];
  __shared__ __bf16 Pt[4][16 * LPT];

  const int tid  = threadIdx.x;
  const int lane = tid & 31;
  const int wave = tid >> 5;
  const int hf = lane >> 4, ln = lane & 15;

  const int bh = blockIdx.y;
  const int b  = bh >> 4, h = bh & (Hh - 1);
  const int qb_block = blockIdx.x * 64;
  const int qb       = qb_block + wave * 16;
  const int q_hi     = qb + 15;

  const __bf16* Qb = Q + (size_t)bh * Tdim * HSd;
  const __bf16* Kb = K + (size_t)bh * Tdim * HSd;
  const __bf16* Vb = V + (size_t)bh * Tdim * HSd;

  // Persistent Q A-frags (HS=64 -> two 32-wide k-steps).
  bf16x16 qf[2];
  {
    const __bf16* qr = Qb + (size_t)(qb + ln) * HSd;
    #pragma unroll
    for (int ks = 0; ks < 2; ++ks)
      qf[ks] = load_frag_split(qr + 32 * ks + 16 * hf,
                               qr + 32 * ks + 16 * hf + 8);
  }

  f32x8 o[4] = {};
  float m_i[8], l_i[8];
  #pragma unroll
  for (int r = 0; r < 8; ++r) { m_i[r] = -__builtin_inff(); l_i[r] = 0.f; }

  // Cooperative staging coordinates: 32 key-rows x 4 16-elem d-chunks.
  const int crow = tid >> 2, cc0 = (tid & 3) * 16;
  const __bf16* kst = Kb + (size_t)crow * HSd + cc0;
  const __bf16* vst = Vb + (size_t)crow * HSd + cc0;

  __bf16 vtmp[16];
#if !HAS_ASYNC
  __bf16 ktmp[16];
#endif

  // Prologue: stage tile kb=0 into buffer 0.
#if HAS_ASYNC
  async_copy_b128(&Kt[0][crow * LKT + cc0],     kst);
  async_copy_b128(&Kt[0][crow * LKT + cc0 + 8], kst + 8);
#else
  __builtin_memcpy(ktmp, kst, 32);
  __builtin_memcpy(&Kt[0][crow * LKT + cc0], ktmp, 32);
#endif
  __builtin_memcpy(vtmp, vst, 32);
  #pragma unroll
  for (int j = 0; j < 16; ++j) Vt[0][(cc0 + j) * LVT + crow] = vtmp[j];

  int cur = 0;
  const int kb_end = qb_block + 63;
  for (int kb = 0; kb <= kb_end; kb += 32) {
#if HAS_ASYNC
    __builtin_amdgcn_s_wait_asynccnt(0);
#endif
    __syncthreads();                           // buf[cur] ready everywhere
    const bool has_next = (kb + 32) <= kb_end;
    if (has_next) {                            // start next tile immediately
#if HAS_ASYNC
      async_copy_b128(&Kt[cur ^ 1][crow * LKT + cc0],     kst + (size_t)(kb + 32) * HSd);
      async_copy_b128(&Kt[cur ^ 1][crow * LKT + cc0 + 8], kst + (size_t)(kb + 32) * HSd + 8);
#else
      __builtin_memcpy(ktmp, kst + (size_t)(kb + 32) * HSd, 32);
#endif
      __builtin_memcpy(vtmp, vst + (size_t)(kb + 32) * HSd, 32);
    }

    if (kb <= q_hi) {
      // ---- S = Q K^T : batch 4 B-frags, then 4 WMMAs --------------------
      const __bf16* Kc = Kt[cur];
      bf16x16 kf0[2], kf1[2];
      #pragma unroll
      for (int ks = 0; ks < 2; ++ks) {
        kf0[ks] = load_frag32(Kc + ln * LKT + 32 * ks + 16 * hf);
        kf1[ks] = load_frag32(Kc + (16 + ln) * LKT + 32 * ks + 16 * hf);
      }
      SCHED_BARRIER();
      f32x8 s0 = {}, s1 = {};
      s0 = wmma_bf16(qf[0], kf0[0], s0);
      s0 = wmma_bf16(qf[1], kf0[1], s0);
      s1 = wmma_bf16(qf[0], kf1[0], s1);
      s1 = wmma_bf16(qf[1], kf1[1], s1);

      // ---- scale, causal mask, online softmax ---------------------------
      float mnew[8];
      #pragma unroll
      for (int r = 0; r < 8; ++r) {
        const int qg = qb + r + 8 * hf;
        float a0 = s0[r] * 0.125f;                     // 1/sqrt(64)
        float a1 = s1[r] * 0.125f;
        if (kb + ln > qg)      a0 = -__builtin_inff();
        if (kb + 16 + ln > qg) a1 = -__builtin_inff();
        s0[r] = a0; s1[r] = a1;
        float v = fmaxf(a0, a1);
        v = fmaxf(v, __shfl_xor(v, 1, 32));
        v = fmaxf(v, __shfl_xor(v, 2, 32));
        v = fmaxf(v, __shfl_xor(v, 4, 32));
        v = fmaxf(v, __shfl_xor(v, 8, 32));            // stays in 16-lane half
        mnew[r] = fmaxf(m_i[r], v);
      }

      #pragma unroll
      for (int r = 0; r < 8; ++r) {
        const float p0 = __expf(s0[r] - mnew[r]);
        const float p1 = __expf(s1[r] - mnew[r]);
        float rs = p0 + p1;
        rs += __shfl_xor(rs, 1, 32);
        rs += __shfl_xor(rs, 2, 32);
        rs += __shfl_xor(rs, 4, 32);
        rs += __shfl_xor(rs, 8, 32);
        const float corr = __expf(m_i[r] - mnew[r]);
        l_i[r] = l_i[r] * corr + rs;
        m_i[r] = mnew[r];
        __bf16* prow = &Pt[wave][(r + 8 * hf) * LPT];
        prow[ln]      = (__bf16)p0;
        prow[16 + ln] = (__bf16)p1;
        #pragma unroll
        for (int nt = 0; nt < 4; ++nt) o[nt][r] *= corr;
      }

      // ---- O += P V : C-layout -> A-layout via per-wave LDS patch -------
      const __bf16* pr = &Pt[wave][ln * LPT];
      bf16x16 pf = load_frag_split(pr + 16 * hf, pr + 16 * hf + 8);
      const __bf16* Vc = Vt[cur];
      bf16x16 vf[4];
      #pragma unroll
      for (int nt = 0; nt < 4; ++nt)
        vf[nt] = load_frag32(Vc + (16 * nt + ln) * LVT + 16 * hf);
      SCHED_BARRIER();
      #pragma unroll
      for (int nt = 0; nt < 4; ++nt)
        o[nt] = wmma_bf16(pf, vf[nt], o[nt]);
    }

    if (has_next) {                            // commit staged V (and K) tile
#if !HAS_ASYNC
      __builtin_memcpy(&Kt[cur ^ 1][crow * LKT + cc0], ktmp, 32);
#endif
      #pragma unroll
      for (int j = 0; j < 16; ++j) Vt[cur ^ 1][(cc0 + j) * LVT + crow] = vtmp[j];
    }
    cur ^= 1;
  }

  // ---- normalize and store y [B, T, H*HS] f32 ---------------------------
  float* yb = Y + (size_t)b * Tdim * (Hh * HSd) + (size_t)h * HSd;
  #pragma unroll
  for (int r = 0; r < 8; ++r) {
    const float inv = 1.0f / l_i[r];
    const int t = qb + r + 8 * hf;
    #pragma unroll
    for (int nt = 0; nt < 4; ++nt)
      yb[(size_t)t * (Hh * HSd) + 16 * nt + ln] = o[nt][r] * inv;
  }
}

// ---------------------------------------------------------------------------
// Kernel 3: output projection  out = y @ wc + bc
// ---------------------------------------------------------------------------
__global__ __launch_bounds__(128) void out_proj_kernel(
    const float* __restrict__ A, const float* __restrict__ W,
    const float* __restrict__ bias, float* __restrict__ out) {
  __shared__ __bf16 As[2][128 * LDA];
  __shared__ __bf16 Bs[2][64 * LDB];

  const int tid  = threadIdx.x;
  const int lane = tid & 31;
  const int wave = tid >> 5;
  const int hf = lane >> 4, ln = lane & 15;

  const int rowBase = blockIdx.x * 128;
  const int colBase = blockIdx.y * 64;

  f32x8 acc[2][4] = {};
  gemm128x64_bf16(A, W, rowBase, colBase, As, Bs, acc);

  #pragma unroll
  for (int rt = 0; rt < 2; ++rt)
    #pragma unroll
    for (int nt = 0; nt < 4; ++nt)
      #pragma unroll
      for (int r = 0; r < 8; ++r) {
        const int row = rowBase + 32 * wave + 16 * rt + r + 8 * hf;
        const int n   = colBase + 16 * nt + ln;
        out[(size_t)row * Cdim + n] = acc[rt][nt][r] + bias[n];
      }
}

// ---------------------------------------------------------------------------
// launch
// ---------------------------------------------------------------------------
extern "C" void kernel_launch(void* const* d_in, const int* in_sizes, int n_in,
                              void* d_out, int out_size, void* d_ws,
                              size_t ws_size, hipStream_t stream) {
  const float* x  = (const float*)d_in[0];
  const float* wq = (const float*)d_in[1];
  const float* bq = (const float*)d_in[2];
  const float* wk = (const float*)d_in[3];
  const float* bk = (const float*)d_in[4];
  const float* wv = (const float*)d_in[5];
  const float* bv = (const float*)d_in[6];
  const float* wc = (const float*)d_in[7];
  const float* bc = (const float*)d_in[8];
  float* out = (float*)d_out;

  const size_t n_qkv = (size_t)Bb * Hh * Tdim * HSd;   // 4,194,304 elems
  __bf16* qws = (__bf16*)d_ws;
  __bf16* kws = qws + n_qkv;
  __bf16* vws = kws + n_qkv;
  float*  yws = (float*)(vws + n_qkv);                 // [B*T, 1024] f32

  qkv_rope_kernel<<<dim3((Bb * Tdim) / 128, 48), 128, 0, stream>>>(
      x, wq, wk, wv, bq, bk, bv, qws, kws, vws);
  attn_kernel<<<dim3(Tdim / 64, Bb * Hh), 128, 0, stream>>>(qws, kws, vws, yws);
  out_proj_kernel<<<dim3((Bb * Tdim) / 128, Cdim / 64), 128, 0, stream>>>(
      yws, wc, bc, out);
}
